// AdditiveAttention_9242769621206
// MI455X (gfx1250) — matrix operand order repaired and checked by reading, not compiled, not run on hardware
//
#include <hip/hip_runtime.h>
#include <hip/hip_bf16.h>

// AdditiveAttention (Bahdanau): B=4, Q=K=512, H=E=DV=256
// out = softmax_k( sum_h w_v[h]*tanh( (q W_q^T)[b,q,h] + (k W_k^T)[b,k,h] ) ) @ values

typedef __attribute__((ext_vector_type(16))) __bf16 v16bf;
typedef __attribute__((ext_vector_type(8)))  __bf16 v8bf;
typedef __attribute__((ext_vector_type(4)))  __bf16 v4bf;
typedef __attribute__((ext_vector_type(8)))  float  v8f;

#define LOG2E 1.4426950408889634f

__device__ __forceinline__ float fast_tanh(float x) {
#if __has_builtin(__builtin_amdgcn_tanhf)
  return __builtin_amdgcn_tanhf(x);          // hardware TANH trans op if exposed
#elif __has_builtin(__builtin_amdgcn_tanh_f32)
  return __builtin_amdgcn_tanh_f32(x);
#else
  // tanh(x) = 1 - 2/(e^{2x}+1); saturates correctly at +-inf
  float t = __builtin_amdgcn_exp2f(x * (2.0f * LOG2E));
  return 1.0f - 2.0f * __builtin_amdgcn_rcpf(t + 1.0f);
#endif
}

// Build 16x32 bf16 A-fragment from two contiguous 8-elt runs:
//   elements 0..7  -> K = half*16 + 0..7       (lo)
//   elements 8..15 -> K = half*16 + 8 + 0..7   (hi)
__device__ __forceinline__ v16bf make_a(v8bf lo, v8bf hi) {
  return __builtin_shufflevector(lo, hi, 0, 1, 2, 3, 4, 5, 6, 7,
                                 8, 9, 10, 11, 12, 13, 14, 15);
}

// ---------------------------------------------------------------------------
// Elementwise f32 -> bf16 (float4 per thread)
// ---------------------------------------------------------------------------
__global__ __launch_bounds__(256) void cvt_bf16(const float* __restrict__ x,
                                                __bf16* __restrict__ y, int n4) {
  int i = blockIdx.x * 256 + threadIdx.x;
  if (i < n4) {
    float4 v = ((const float4*)x)[i];
    v4bf o;
    o[0] = (__bf16)v.x; o[1] = (__bf16)v.y; o[2] = (__bf16)v.z; o[3] = (__bf16)v.w;
    ((v4bf*)y)[i] = o;
  }
}

// values [4,512,256] f32 -> vt [4,256,512] bf16 (coalesced reads)
__global__ __launch_bounds__(256) void transpose_cvt_values(const float* __restrict__ v,
                                                            __bf16* __restrict__ vt) {
  int i = blockIdx.x * 256 + threadIdx.x;      // 0 .. 4*512*256-1
  int d = i & 255;
  int k = (i >> 8) & 511;
  int b = i >> 17;
  vt[((size_t)(b * 256 + d)) * 512 + k] = (__bf16)v[i];
}

// ---------------------------------------------------------------------------
// Projection GEMM: C[m,h] = sum_e X[m,e]*W[h,e], X/W bf16, C f32.
// One wave -> 16x64 C tile (4 accumulators, A reused x4). E in steps of 32.
// A frag: lane -> row m0+l%16, half=l/16; K = half*16+e via two 8-bf16 runs (16B loads).
// B frag: lane -> col, elements e -> K = half*16+e: one 32B contiguous load.
// ---------------------------------------------------------------------------
template<int E_, int H_>
__global__ __launch_bounds__(256) void proj_wmma(const __bf16* __restrict__ X,
                                                 const __bf16* __restrict__ W,
                                                 float* __restrict__ C) {
  const int lane = threadIdx.x & 31;
  const int wave = threadIdx.x >> 5;
  const int tile = blockIdx.x * 8 + wave;      // (M/16)*(H/64) tiles
  constexpr int tilesN = H_ / 64;
  const int m0 = (tile / tilesN) * 16;
  const int n0 = (tile % tilesN) * 64;
  const int half = lane >> 4;
  const int mr = m0 + (lane & 15);
  const int hc = n0 + (lane & 15);

  const __bf16* xr = X + (size_t)mr * E_;
  v8f acc[4] = {};

#pragma unroll
  for (int e0 = 0; e0 < E_; e0 += 32) {
    v8bf alo = *(const v8bf*)(xr + e0 + half * 16);
    v8bf ahi = *(const v8bf*)(xr + e0 + half * 16 + 8);
    v16bf a = make_a(alo, ahi);
#pragma unroll
    for (int s = 0; s < 4; ++s) {
      const __bf16* wb = W + (size_t)(hc + s * 16) * E_ + e0 + half * 16;
      v16bf b = *(const v16bf*)wb;
      acc[s] = __builtin_amdgcn_wmma_f32_16x16x32_bf16(false, a, false, b,
                                                       (short)0, acc[s], false, false);
    }
  }
  // C/D layout: VGPR r, lane l -> row = m0 + half*8 + r, col = l%16 (+s*16)
#pragma unroll
  for (int s = 0; s < 4; ++s) {
    float* cr = C + (size_t)(m0 + half * 8) * H_ + hc + s * 16;
#pragma unroll
    for (int r = 0; r < 8; ++r) cr[(size_t)r * H_] = acc[s][r];
  }
}

// ---------------------------------------------------------------------------
// Scores + softmax. One 256-thread block per (b,q) row.
// Lane owns h-chunk [lane*8, lane*8+8) in registers; a wave sweeps k with
// coalesced row reads; __shfl_xor wave reduction -> score; LDS softmax.
// Emits attn in bf16 (PV GEMM A operand).
// ---------------------------------------------------------------------------
__global__ __launch_bounds__(256) void scores_softmax(const float* __restrict__ qp,
                                                      const float* __restrict__ kp,
                                                      const float* __restrict__ wv,
                                                      __bf16* __restrict__ attn) {
  constexpr int Hd = 256, Kn = 512, Qn = 512;
  __shared__ float s_s[Kn];
  __shared__ float s_red[256];
  const int t = threadIdx.x;
  const int lane = t & 31;
  const int wave = t >> 5;
  const int bq = blockIdx.x;
  const int b = bq / Qn;
  const int hbase = lane * 8;

  // q-row chunk + w_v chunk in registers (loaded once)
  float qv[8], wvv[8];
  {
    const float* qr = qp + (size_t)bq * Hd + hbase;
    float4 q0 = *(const float4*)(qr);
    float4 q1 = *(const float4*)(qr + 4);
    qv[0] = q0.x; qv[1] = q0.y; qv[2] = q0.z; qv[3] = q0.w;
    qv[4] = q1.x; qv[5] = q1.y; qv[6] = q1.z; qv[7] = q1.w;
    const float* wr = wv + hbase;
    float4 w0 = *(const float4*)(wr);
    float4 w1 = *(const float4*)(wr + 4);
    wvv[0] = w0.x; wvv[1] = w0.y; wvv[2] = w0.z; wvv[3] = w0.w;
    wvv[4] = w1.x; wvv[5] = w1.y; wvv[6] = w1.z; wvv[7] = w1.w;
  }

  const float* kbase = kp + (size_t)b * Kn * Hd + hbase;
  for (int k = wave; k < Kn; k += 8) {
    if (k + 64 < Kn) __builtin_prefetch(kbase + (size_t)(k + 64) * Hd, 0, 1);
    const float* kr = kbase + (size_t)k * Hd;
    float4 k0 = *(const float4*)(kr);
    float4 k1 = *(const float4*)(kr + 4);
    float s = wvv[0] * fast_tanh(qv[0] + k0.x);
    s += wvv[1] * fast_tanh(qv[1] + k0.y);
    s += wvv[2] * fast_tanh(qv[2] + k0.z);
    s += wvv[3] * fast_tanh(qv[3] + k0.w);
    s += wvv[4] * fast_tanh(qv[4] + k1.x);
    s += wvv[5] * fast_tanh(qv[5] + k1.y);
    s += wvv[6] * fast_tanh(qv[6] + k1.z);
    s += wvv[7] * fast_tanh(qv[7] + k1.w);
#pragma unroll
    for (int off = 16; off > 0; off >>= 1) s += __shfl_xor(s, off, 32);
    if (lane == 0) s_s[k] = s;
  }
  __syncthreads();

  // softmax over 512 scores
  s_red[t] = fmaxf(s_s[t], s_s[t + 256]);
  __syncthreads();
  for (int off = 128; off > 0; off >>= 1) {
    if (t < off) s_red[t] = fmaxf(s_red[t], s_red[t + off]);
    __syncthreads();
  }
  const float rowmax = s_red[0];
  __syncthreads();
  const float e0 = __builtin_amdgcn_exp2f((s_s[t] - rowmax) * LOG2E);
  const float e1 = __builtin_amdgcn_exp2f((s_s[t + 256] - rowmax) * LOG2E);
  s_red[t] = e0 + e1;
  __syncthreads();
  for (int off = 128; off > 0; off >>= 1) {
    if (t < off) s_red[t] += s_red[t + off];
    __syncthreads();
  }
  const float inv = __builtin_amdgcn_rcpf(s_red[0]);

  __bf16* ar = attn + (size_t)bq * Kn;
  ar[t] = (__bf16)(e0 * inv);
  ar[t + 256] = (__bf16)(e1 * inv);
}

// ---------------------------------------------------------------------------
// PV GEMM: out[b,q,d] = sum_k attn[b,q,k]*values[b,k,d]
// attn bf16 [b][q][k]; vt bf16 [b][d][k] (transposed) -> B frag is one 32B load.
// One wave -> 16x64 out tile (4 accumulators, A reused x4).
// ---------------------------------------------------------------------------
__global__ __launch_bounds__(256) void pv_wmma(const __bf16* __restrict__ attn,
                                               const __bf16* __restrict__ vt,
                                               float* __restrict__ out) {
  constexpr int Qn = 512, Kn = 512, DVn = 256;
  const int lane = threadIdx.x & 31;
  const int wave = threadIdx.x >> 5;
  const int tile = blockIdx.x * 8 + wave;            // 4*(512/16)*(256/64)=512
  constexpr int tilesN = DVn / 64;                   // 4
  constexpr int tilesPerB = (Qn / 16) * tilesN;      // 128
  const int b = tile / tilesPerB;
  const int rem = tile % tilesPerB;
  const int q0 = (rem / tilesN) * 16;
  const int d0 = (rem % tilesN) * 64;
  const int half = lane >> 4;
  const int qr = q0 + (lane & 15);
  const int dc = d0 + (lane & 15);

  const __bf16* arow = attn + ((size_t)b * Qn + qr) * Kn;
  const __bf16* vbb = vt + (size_t)b * DVn * Kn;

  v8f acc[4] = {};
#pragma unroll 4
  for (int k0 = 0; k0 < Kn; k0 += 32) {
    v8bf alo = *(const v8bf*)(arow + k0 + half * 16);
    v8bf ahi = *(const v8bf*)(arow + k0 + half * 16 + 8);
    v16bf a = make_a(alo, ahi);
#pragma unroll
    for (int s = 0; s < 4; ++s) {
      const __bf16* vb = vbb + (size_t)(dc + s * 16) * Kn + k0 + half * 16;
      v16bf bv = *(const v16bf*)vb;
      acc[s] = __builtin_amdgcn_wmma_f32_16x16x32_bf16(false, a, false, bv,
                                                       (short)0, acc[s], false, false);
    }
  }
#pragma unroll
  for (int s = 0; s < 4; ++s) {
    float* orow = out + ((size_t)b * Qn + q0 + half * 8) * DVn + dc + s * 16;
#pragma unroll
    for (int r = 0; r < 8; ++r) orow[(size_t)r * DVn] = acc[s][r];
  }
}

// ---------------------------------------------------------------------------
extern "C" void kernel_launch(void* const* d_in, const int* in_sizes, int n_in,
                              void* d_out, int out_size, void* d_ws, size_t ws_size,
                              hipStream_t stream) {
  const float* queries = (const float*)d_in[0];  // [4,512,256]
  const float* keys    = (const float*)d_in[1];  // [4,512,256]
  const float* values  = (const float*)d_in[2];  // [4,512,256]
  const float* W_q     = (const float*)d_in[3];  // [256,256]
  const float* W_k     = (const float*)d_in[4];  // [256,256]
  const float* w_v     = (const float*)d_in[5];  // [256]
  float* out = (float*)d_out;                    // [4,512,256]

  // workspace carve (all regions overwritten before read; sizes in bytes)
  char* p = (char*)d_ws;
  __bf16* qb    = (__bf16*)p; p += (size_t)2048 * 256 * 2;   // 1 MB
  __bf16* kb    = (__bf16*)p; p += (size_t)2048 * 256 * 2;   // 1 MB
  __bf16* wqb   = (__bf16*)p; p += (size_t)256 * 256 * 2;    // 128 KB
  __bf16* wkb   = (__bf16*)p; p += (size_t)256 * 256 * 2;    // 128 KB
  __bf16* vtb   = (__bf16*)p; p += (size_t)4 * 256 * 512 * 2;// 1 MB
  __bf16* attnb = (__bf16*)p; p += (size_t)4 * 512 * 512 * 2;// 2 MB
  float*  qp    = (float*)p;  p += (size_t)2048 * 256 * 4;   // 2 MB
  float*  kp    = (float*)p;                                  // 2 MB

  // stage 0: precision/layout conversions (independent)
  cvt_bf16<<<512, 256, 0, stream>>>(queries, qb, 2048 * 256 / 4);
  cvt_bf16<<<512, 256, 0, stream>>>(keys, kb, 2048 * 256 / 4);
  cvt_bf16<<<64, 256, 0, stream>>>(W_q, wqb, 256 * 256 / 4);
  cvt_bf16<<<64, 256, 0, stream>>>(W_k, wkb, 256 * 256 / 4);
  transpose_cvt_values<<<2048, 256, 0, stream>>>(values, vtb);

  // stage 1: projections (512 waves each -> 64 blocks of 8 waves)
  proj_wmma<256, 256><<<64, 256, 0, stream>>>(qb, wqb, qp);
  proj_wmma<256, 256><<<64, 256, 0, stream>>>(kb, wkb, kp);

  // stage 2: scores + softmax (one block per (b,q) row)
  scores_softmax<<<2048, 256, 0, stream>>>(qp, kp, w_v, attnb);

  // stage 3: attn @ values (512 waves -> 64 blocks)
  pv_wmma<<<64, 256, 0, stream>>>(attnb, vtb, out);
}
